// CKMachine_70025146794650
// MI455X (gfx1250) — matrix operand order repaired and checked by reading, not compiled, not run on hardware
//
#include <hip/hip_runtime.h>
#include <hip/hip_bf16.h>
#include <math.h>
#include <stdint.h>

// Problem constants (from the reference)
#define S_   1024
#define B_   2048
#define N_   3072          // S + B
#define C_   8
#define F_   10
#define D_   1352          // 13*13*8
#define DP_  1408          // D padded to multiple of 64 (GEMM tile)
#define TAU_ 6.28318530717958647692f

typedef __attribute__((ext_vector_type(8)))  __bf16 v8bf;
typedef __attribute__((ext_vector_type(16))) __bf16 v16bf;
typedef __attribute__((ext_vector_type(8)))  float  v8f;

// ---- CDNA5 async global->LDS copy (ASYNCcnt-tracked, no VGPR round-trip)
__device__ __forceinline__ void async_copy16(const void* gsrc, void* ldst) {
    unsigned loff = (unsigned)(unsigned long long)(uintptr_t)ldst; // LDS byte offset
    asm volatile("global_load_async_to_lds_b128 %0, %1, off"
                 :: "v"(loff), "v"(gsrc) : "memory");
}
__device__ __forceinline__ void async_wait0() {
    asm volatile("s_wait_asynccnt 0x0" ::: "memory");
}

// ---------------------------------------------------------------- elementwise
__global__ void k_zero(float* p, long n) {
    long i = (long)blockIdx.x * blockDim.x + threadIdx.x;
    if (i < n) p[i] = 0.0f;
}

__global__ void k_tobf16(const float* __restrict__ s, __bf16* __restrict__ d, long n) {
    long i = (long)blockIdx.x * blockDim.x + threadIdx.x;
    if (i < n) d[i] = (__bf16)s[i];
}

// dst = a + alpha*b
__global__ void k_fma(float* __restrict__ dst, const float* __restrict__ a,
                      const float* __restrict__ b, float alpha, long n) {
    long i = (long)blockIdx.x * blockDim.x + threadIdx.x;
    if (i < n) dst[i] = a[i] + alpha * b[i];
}

// acc += coef*k
__global__ void k_acc(float* __restrict__ acc, const float* __restrict__ kk,
                      float coef, long n) {
    long i = (long)blockIdx.x * blockDim.x + threadIdx.x;
    if (i < n) acc[i] += coef * kk[i];
}

// ----------------------------------------------- per-time weight combination
// wbuf[0..71]=wcomb[c][ky][kx], wbuf[72..79]=bcomb[c], wbuf[80..89]=cos(t*TAU*f)
__global__ void k_wcomb(const float* __restrict__ conv_w, const float* __restrict__ conv_b,
                        float* __restrict__ wbuf, float t) {
    int tid = threadIdx.x;
    if (tid < 72) {
        int c = tid / 9, kk = tid % 9;
        float a = 0.f;
        #pragma unroll
        for (int f = 0; f < F_; ++f) a += cosf(t * TAU_ * f) * conv_w[(f * C_ + c) * 9 + kk];
        wbuf[tid] = a;
    } else if (tid < 80) {
        int c = tid - 72;
        float a = 0.f;
        #pragma unroll
        for (int f = 0; f < F_; ++f) a += cosf(t * TAU_ * f) * conv_b[f * C_ + c];
        wbuf[tid] = a;
    } else if (tid < 90) {
        wbuf[tid] = cosf(t * TAU_ * (tid - 80));
    }
}

// ------------------------------------------ psi_t[n][d] = conv(x)·w  (fp32, padded)
__global__ void k_psit(const float* __restrict__ data, const float* __restrict__ inputs,
                       const float* __restrict__ wbuf, float* __restrict__ psi) {
    long idx = (long)blockIdx.x * blockDim.x + threadIdx.x;
    if (idx >= (long)N_ * DP_) return;
    int n = (int)(idx / DP_), d = (int)(idx % DP_);
    float val = 0.f;
    if (d < D_) {
        int c = d / 169, rem = d % 169, h = rem / 13, w = rem % 13;
        const float* img = (n < S_) ? (data + (size_t)n * 784)
                                    : (inputs + (size_t)(n - S_) * 784);
        const float* wc = wbuf + c * 9;
        float acc = wbuf[72 + c];
        int r0 = 2 * h, c0 = 2 * w;
        #pragma unroll
        for (int ky = 0; ky < 3; ++ky)
            #pragma unroll
            for (int kx = 0; kx < 3; ++kx)
                acc += img[(r0 + ky) * 28 + (c0 + kx)] * wc[ky * 3 + kx];
        val = acc;
    }
    psi[idx] = val;
}

// ------------------------------ c_t TRANSPOSED: ctT[d][s] = c_s[s,d,:]·w  (bf16 [+f32])
__global__ void k_ct(const float* __restrict__ c_s, const float* __restrict__ wbuf,
                     __bf16* __restrict__ ctT, float* __restrict__ ctTf, int wantF32) {
    long idx = (long)blockIdx.x * blockDim.x + threadIdx.x;
    if (idx >= (long)DP_ * S_) return;
    int d = (int)(idx / S_), s = (int)(idx % S_);
    float v = 0.f;
    if (d < D_) {
        const float* w  = wbuf + 80;
        const float* cs = c_s + ((size_t)s * D_ + d) * F_;
        #pragma unroll
        for (int f = 0; f < F_; ++f) v += cs[f] * w[f];
    }
    ctT[idx] = (__bf16)v;
    if (wantF32) ctTf[idx] = v;
}

// ------------------------------------------------------------------- squash
__global__ __launch_bounds__(256) void k_squash(const float* __restrict__ u,
                                                __bf16* __restrict__ vbf,
                                                float* __restrict__ uu) {
    __shared__ float red[256];
    int row = blockIdx.x, tid = threadIdx.x;
    const float* ur = u + (size_t)row * DP_;
    float s = 0.f;
    for (int d = tid; d < DP_; d += 256) { float x = ur[d]; s += x * x; }
    red[tid] = s; __syncthreads();
    for (int o = 128; o > 0; o >>= 1) { if (tid < o) red[tid] += red[tid + o]; __syncthreads(); }
    float sum   = red[0];
    float norm  = sqrtf(sum + 1e-6f);
    float scale = (1.f / (1.f + expf(-norm))) / norm;
    __bf16* vr = vbf + (size_t)row * DP_;
    for (int d = tid; d < DP_; d += 256) vr[d] = (__bf16)(ur[d] * scale);
    if (tid == 0) uu[row] = sum * scale * scale;   // ||v||^2
}

// --------------------------------------------------------------- record z_t
__global__ void k_record(const float* __restrict__ z, float* __restrict__ zsA,
                         float* __restrict__ out, int tslot) {
    long idx = (long)blockIdx.x * blockDim.x + threadIdx.x;
    if (idx >= (long)N_ * S_) return;
    int n = (int)(idx / S_), s2 = (int)(idx % S_);
    float v = z[idx];
    if (n < S_) zsA[((size_t)tslot * S_ + n) * S_ + s2] = v;
    else        out[((size_t)(n - S_) * F_ + tslot) * S_ + s2] = v;
}

// ------------------------------------------------------------ WMMA bf16 GEMM
// D[m][n] = sum_k A[m][k] * B[n][k]   (B stored n-major, contiguous K)
// Block 256 thr = 8 waves (4x2), tile 128x64, K-step 64, double-buffered LDS.
// Async copies for tile i+1 are issued right after the barrier, then tile i is
// consumed -> copy/compute overlap; one barrier per K-step.
// Each wave: 32x32 output, 8x v_wmma_f32_16x16x32_bf16 per K-step.
// mode 0: D = acc + aux0[m*ldd+n]                     (u = z@c + psi)
// mode 1: D = exp(acc - aux0[m]/2 - aux0[n]/2) * acc  (adjusted kernel; vv==uu[:S])
// mode 2: D = acc
#define KSTEP 64
__global__ __launch_bounds__(256)
void k_gemm(const __bf16* __restrict__ A, int lda,
            const __bf16* __restrict__ B, int ldb,
            float* __restrict__ Dst, int ldd,
            const float* __restrict__ aux0, int K, int mode) {
    __shared__ __align__(16) __bf16 sA[2][128][KSTEP];   // 2 x 16KB
    __shared__ __align__(16) __bf16 sB[2][ 64][KSTEP];   // 2 x  8KB
    const int tid  = threadIdx.x;
    const int lane = tid & 31;
    const int wv   = tid >> 5;            // 0..7
    const int wm   = wv & 3;              // 32-row subtile
    const int wn   = wv >> 2;             // 32-col subtile
    const int Mb   = blockIdx.x * 128;
    const int Nb   = blockIdx.y * 64;

    const int ar = tid >> 1, ac = (tid & 1) * 32;   // A: 128 rows x 2 chunks of 32 halves
    const int br = tid >> 2, bc = (tid & 3) * 16;   // B: 64 rows x 4 chunks of 16 halves
    const int hlf = lane >> 4, r16 = lane & 15;

    const __bf16* Abase = A + (size_t)(Mb + ar) * lda + ac;
    const __bf16* Bbase = B + (size_t)(Nb + br) * ldb + bc;

    auto issue_tile = [&](int k0, int buf) {
        const __bf16* Ap = Abase + k0;
        async_copy16(Ap,      &sA[buf][ar][ac]);
        async_copy16(Ap + 8,  &sA[buf][ar][ac + 8]);
        async_copy16(Ap + 16, &sA[buf][ar][ac + 16]);
        async_copy16(Ap + 24, &sA[buf][ar][ac + 24]);
        const __bf16* Bp = Bbase + k0;
        async_copy16(Bp,      &sB[buf][br][bc]);
        async_copy16(Bp + 8,  &sB[buf][br][bc + 8]);
    };

    v8f acc[2][2] = {};
    int cur = 0;
    issue_tile(0, 0);

    for (int k0 = 0; k0 < K; k0 += KSTEP) {
        async_wait0();        // copies into sX[cur] complete (this wave)
        __syncthreads();      // everyone's copies done; prev tile fully consumed
        if (k0 + KSTEP < K) issue_tile(k0 + KSTEP, cur ^ 1);   // overlap with compute

        #pragma unroll
        for (int kc = 0; kc < KSTEP; kc += 32) {
            // A fragments (16x32): lane half -> K {h*8..h*8+7} U {16+h*8..}
            v16bf af[2];
            #pragma unroll
            for (int mi = 0; mi < 2; ++mi) {
                int rr = wm * 32 + mi * 16 + r16;
                v8bf lo = *(const v8bf*)(&sA[cur][rr][kc + hlf * 8]);
                v8bf hi = *(const v8bf*)(&sA[cur][rr][kc + 16 + hlf * 8]);
                #pragma unroll
                for (int j = 0; j < 8; ++j) { af[mi][j] = lo[j]; af[mi][8 + j] = hi[j]; }
            }
            // B fragments (32x16): lane half -> K {h*8..h*8+7} U {16+h*8..}
            v16bf bfr[2];
            #pragma unroll
            for (int ni = 0; ni < 2; ++ni) {
                int nc = wn * 32 + ni * 16 + r16;
                v8bf lo = *(const v8bf*)(&sB[cur][nc][kc + hlf * 8]);
                v8bf hi = *(const v8bf*)(&sB[cur][nc][kc + 16 + hlf * 8]);
                #pragma unroll
                for (int j = 0; j < 8; ++j) { bfr[ni][j] = lo[j]; bfr[ni][8 + j] = hi[j]; }
            }
            #pragma unroll
            for (int mi = 0; mi < 2; ++mi)
                #pragma unroll
                for (int ni = 0; ni < 2; ++ni)
                    acc[mi][ni] = __builtin_amdgcn_wmma_f32_16x16x32_bf16(
                        false, af[mi], false, bfr[ni], (short)0, acc[mi][ni], false, false);
        }
        cur ^= 1;
    }

    // ---- epilogue: D layout (VGPR r -> M=r+8*half, N=lane&15)
    #pragma unroll
    for (int mi = 0; mi < 2; ++mi) {
        const int m0 = Mb + wm * 32 + mi * 16 + hlf * 8;
        #pragma unroll
        for (int ni = 0; ni < 2; ++ni) {
            const int n0 = Nb + wn * 32 + ni * 16 + r16;
            #pragma unroll
            for (int r = 0; r < 8; ++r) {
                int m = m0 + r;
                float x = acc[mi][ni][r];
                if (mode == 0) {
                    x += aux0[(size_t)m * ldd + n0];
                } else if (mode == 1) {
                    x = expf(x - 0.5f * aux0[m] - 0.5f * aux0[n0]) * x;
                }
                Dst[(size_t)m * ldd + n0] = x;
            }
        }
    }
}

// ------------------------------------------------------------------- cost
__global__ void k_costinit(const float* __restrict__ cw, const float* __restrict__ cb,
                           float* __restrict__ cost) {
    __shared__ float red[256];
    int tid = threadIdx.x;
    float s = 0.f;
    for (int i = tid; i < 800; i += 256) {
        float v = (i < 720) ? cw[i] : cb[i - 720];
        s += v * v;
    }
    red[tid] = s; __syncthreads();
    for (int o = 128; o > 0; o >>= 1) { if (tid < o) red[tid] += red[tid + o]; __syncthreads(); }
    if (tid == 0) cost[0] = red[0];
}

// sum_{i,d} y[i][d] * ctT[d][i]   (y row-major [S][Dp], ctT [Dp][S])
__global__ __launch_bounds__(256) void k_dot(const float* __restrict__ y,
                                             const float* __restrict__ ctTf,
                                             float* __restrict__ part, long n) {
    __shared__ float red[256];
    int tid = threadIdx.x;
    float s = 0.f;
    for (long idx = (long)blockIdx.x * 256 + tid; idx < n; idx += (long)gridDim.x * 256) {
        int i = (int)(idx / DP_), d = (int)(idx % DP_);
        s += y[idx] * ctTf[(size_t)d * S_ + i];
    }
    red[tid] = s; __syncthreads();
    for (int o = 128; o > 0; o >>= 1) { if (tid < o) red[tid] += red[tid + o]; __syncthreads(); }
    if (tid == 0) part[blockIdx.x] = red[0];
}

__global__ void k_fin(const float* __restrict__ part, float* __restrict__ cost, float scale) {
    __shared__ float red[256];
    int tid = threadIdx.x;
    float s = part[tid] + part[tid + 256] + part[tid + 512] + part[tid + 768];
    red[tid] = s; __syncthreads();
    for (int o = 128; o > 0; o >>= 1) { if (tid < o) red[tid] += red[tid + o]; __syncthreads(); }
    if (tid == 0) cost[0] += red[0] * scale;
}

// ================================================================== launch
extern "C" void kernel_launch(void* const* d_in, const int* in_sizes, int n_in,
                              void* d_out, int out_size, void* d_ws, size_t ws_size,
                              hipStream_t stream) {
    const float* data   = (const float*)d_in[0];
    const float* inputs = (const float*)d_in[1];
    const float* conv_w = (const float*)d_in[2];
    const float* conv_b = (const float*)d_in[3];
    const float* c_s    = (const float*)d_in[4];
    float* out = (float*)d_out;
    (void)in_sizes; (void)n_in; (void)out_size; (void)ws_size;

    // ---- workspace carve-up (256B aligned)
    char* base = (char*)d_ws;
    size_t off = 0;
    auto alloc = [&](size_t bytes) {
        off = (off + 255) & ~(size_t)255;
        size_t o = off; off += bytes; return (void*)(base + o);
    };
    const long NS  = (long)N_ * S_;
    const long NDp = (long)N_ * DP_;
    const long SDp = (long)S_ * DP_;

    float*  z    = (float*) alloc(NS  * 4);
    float*  ztmp = (float*) alloc(NS  * 4);
    float*  kbuf = (float*) alloc(NS  * 4);
    float*  kacc = (float*) alloc(NS  * 4);
    __bf16* zbf  = (__bf16*)alloc(NS  * 2);
    __bf16* ctT  = (__bf16*)alloc(SDp * 2);   // [Dp][S] bf16
    float*  ctTf = (float*) alloc(SDp * 4);   // [Dp][S] f32 (cost)
    float*  psi  = (float*) alloc(NDp * 4);
    float*  u    = (float*) alloc(NDp * 4);   // reused as y in cost phase
    __bf16* vbf  = (__bf16*)alloc(NDp * 2);
    float*  uu   = (float*) alloc(N_  * 4);
    float*  wbuf = (float*) alloc(128 * 4);
    float*  zsA  = (float*) alloc((size_t)F_ * S_ * S_ * 4);
    float*  part = (float*) alloc(1024 * 4);

    auto blk = [](long n) { return dim3((unsigned)((n + 255) / 256)); };

    // ---- one RHS evaluation: kbuf = f(t, zin)
    auto stage = [&](float t, const float* zin) {
        k_wcomb<<<1, 128, 0, stream>>>(conv_w, conv_b, wbuf, t);
        k_psit <<<blk(NDp), 256, 0, stream>>>(data, inputs, wbuf, psi);
        k_ct   <<<blk(SDp), 256, 0, stream>>>(c_s, wbuf, ctT, nullptr, 0);
        k_tobf16<<<blk(NS), 256, 0, stream>>>(zin, zbf, NS);
        dim3 g1(N_ / 128, DP_ / 64);                     // u = z@c + psi
        k_gemm<<<g1, 256, 0, stream>>>(zbf, S_, ctT, S_, u, DP_, psi, S_, 0);
        k_squash<<<N_, 256, 0, stream>>>(u, vbf, uu);
        dim3 g2(N_ / 128, S_ / 64);                      // k = kernel(v, v[:S])
        k_gemm<<<g2, 256, 0, stream>>>(vbf, DP_, vbf, DP_, kbuf, S_, uu, DP_, 1);
    };

    // ---- init
    k_zero<<<blk(NS), 256, 0, stream>>>(z, NS);
    k_zero<<<blk((long)B_ * F_ * S_ + 1), 256, 0, stream>>>(out, (long)B_ * F_ * S_ + 1);
    k_zero<<<blk((long)F_ * S_ * S_), 256, 0, stream>>>(zsA, (long)F_ * S_ * S_);

    // ---- RK4 trajectory (9 steps, h = 1/9)
    const float h = 1.0f / 9.0f;
    for (int i = 0; i < 9; ++i) {
        float t0 = i * h;
        k_zero<<<blk(NS), 256, 0, stream>>>(kacc, NS);
        stage(t0, z);                                               // k1
        k_acc<<<blk(NS), 256, 0, stream>>>(kacc, kbuf, 1.f, NS);
        k_fma<<<blk(NS), 256, 0, stream>>>(ztmp, z, kbuf, h * 0.5f, NS);
        stage(t0 + h * 0.5f, ztmp);                                 // k2
        k_acc<<<blk(NS), 256, 0, stream>>>(kacc, kbuf, 2.f, NS);
        k_fma<<<blk(NS), 256, 0, stream>>>(ztmp, z, kbuf, h * 0.5f, NS);
        stage(t0 + h * 0.5f, ztmp);                                 // k3
        k_acc<<<blk(NS), 256, 0, stream>>>(kacc, kbuf, 2.f, NS);
        k_fma<<<blk(NS), 256, 0, stream>>>(ztmp, z, kbuf, h, NS);
        stage(t0 + h, ztmp);                                        // k4
        k_acc<<<blk(NS), 256, 0, stream>>>(kacc, kbuf, 1.f, NS);
        k_fma<<<blk(NS), 256, 0, stream>>>(z, z, kacc, h / 6.f, NS);
        k_record<<<blk(NS), 256, 0, stream>>>(z, zsA, out, i + 1);
    }

    // ---- cost = ||conv_w||^2 + ||conv_b||^2 + (1/F) sum_t <z_t[:S]@c_t , c_t>
    float* cost = out + (size_t)B_ * F_ * S_;
    k_costinit<<<1, 256, 0, stream>>>(conv_w, conv_b, cost);
    for (int tt = 0; tt < F_; ++tt) {
        float t = tt / 9.0f;
        k_wcomb<<<1, 128, 0, stream>>>(conv_w, conv_b, wbuf, t);
        k_ct  <<<blk(SDp), 256, 0, stream>>>(c_s, wbuf, ctT, ctTf, 1);
        k_tobf16<<<blk((long)S_ * S_), 256, 0, stream>>>(zsA + (size_t)tt * S_ * S_, zbf,
                                                         (long)S_ * S_);
        dim3 gc(S_ / 128, DP_ / 64);                     // y = z_t[:S] @ c_t
        k_gemm<<<gc, 256, 0, stream>>>(zbf, S_, ctT, S_, u, DP_, nullptr, S_, 2);
        k_dot<<<1024, 256, 0, stream>>>(u, ctTf, part, SDp);
        k_fin<<<1, 256, 0, stream>>>(part, cost, 1.0f / F_);
    }
}
